// ReacFullGbModel_69269232550276
// MI455X (gfx1250) — matrix operand order repaired and checked by reading, not compiled, not run on hardware
//
#include <hip/hip_runtime.h>
#include <hip/hip_bf16.h>

typedef __attribute__((ext_vector_type(16))) _Float16     v16h;
typedef __attribute__((ext_vector_type(2)))  __fp16       v2hf;   // return type of cvt_pkrtz
typedef __attribute__((ext_vector_type(8)))  float        v8f;
typedef __attribute__((ext_vector_type(8)))  unsigned int v8u;

#define T_STEPS 1024
#define B_ELEMS 4096
#define NXZ     18     // Nx + Nz = 3 + 15

__device__ __forceinline__ float fast_tanh(float x) {
#if __has_builtin(__builtin_amdgcn_tanhf)
  return __builtin_amdgcn_tanhf(x);
#else
  return tanhf(x);
#endif
}

// pack two f32 into one dword of f16 pairs (low = a, high = b)
__device__ __forceinline__ unsigned int pk2h(float a, float b) {
#if __has_builtin(__builtin_amdgcn_cvt_pkrtz)
  v2hf p = __builtin_amdgcn_cvt_pkrtz(a, b);
  return __builtin_bit_cast(unsigned int, p);
#else
  unsigned int lo = (unsigned int)__builtin_bit_cast(unsigned short, (_Float16)a);
  unsigned int hi = (unsigned int)__builtin_bit_cast(unsigned short, (_Float16)b);
  return lo | (hi << 16);
#endif
}

__global__ __launch_bounds__(32, 1)
void reac_scan_kernel(const float* __restrict__ u,
                      const float* __restrict__ xz0,
                      const float* __restrict__ r1_W1, const float* __restrict__ r1_b1,
                      const float* __restrict__ r1_W2, const float* __restrict__ r1_b2,
                      const float* __restrict__ r2_W1, const float* __restrict__ r2_b1,
                      const float* __restrict__ r2_W2, const float* __restrict__ r2_b2,
                      const float* __restrict__ eC_W1, const float* __restrict__ eC_b1,
                      const float* __restrict__ eC_W2, const float* __restrict__ eC_b2,
                      float* __restrict__ out)
{
  const int lane = threadIdx.x;            // block = exactly one wave32
  const int b    = blockIdx.x * 32 + lane; // batch element owned by this lane
  const bool lolane = (lane < 16);
  const int m = lane & 15;

  // ---------------- per-lane weight registers (broadcast loads) -------------
  float r1w1[16], r1b1v[16], r1w2[16];
  float r2w1a[16], r2w1b[16], r2b1v[16], r2w2[16];
#pragma unroll
  for (int h = 0; h < 16; ++h) {
    r1w1[h]  = r1_W1[h];        // (1,16)
    r1b1v[h] = r1_b1[h];
    r1w2[h]  = r1_W2[h];        // (16,1)
    r2w1a[h] = r2_W1[h];        // (2,16): row 0
    r2w1b[h] = r2_W1[16 + h];   //          row 1
    r2b1v[h] = r2_b1[h];
    r2w2[h]  = r2_W2[h];        // (16,1)
  }
  const float r1b2v = r1_b2[0];
  const float r2b2v = r2_b2[0];

  // ---------------- eC WMMA fragments --------------------------------------
  // D = A x B with A = W1^T tile (M=hidden 16, K=z-dim padded to 32),
  //                B = Z^T tile  (K=z-dim, N=element 16).
  // 16-bit A 16x32 layout (contiguous, matching B): lanes 0-15 (M=lane): half i -> K = i
  //                                     lanes 16-31 (M=lane-16): K = i+16
  v16h A0 = {}, A1 = {};   // hidden tiles h = 0..15 and 16..31
#pragma unroll
  for (int i = 0; i < 16; ++i) {
    int k = lolane ? i
                   : i + 16;
    float w0 = 0.f, w1 = 0.f;
    if (k < 15) {                       // Nz = 15; K >= 15 is zero padding
      w0 = eC_W1[k * 32 + m];           // W1 row-major (15,32)
      w1 = eC_W1[k * 32 + 16 + m];
    }
    A0[i] = (_Float16)w0;
    A1[i] = (_Float16)w1;
  }
  // layer-2 constants: lane handles hidden rows h = r + hoff and h + 16
  const int hoff = lolane ? 0 : 8;
  float eb1A[8], eb1B[8], ew2A[8], ew2B[8];
#pragma unroll
  for (int r = 0; r < 8; ++r) {
    eb1A[r] = eC_b1[hoff + r];
    ew2A[r] = eC_W2[hoff + r];
    eb1B[r] = eC_b1[hoff + r + 16];
    ew2B[r] = eC_W2[hoff + r + 16];
  }
  const float eb2 = eC_b2[0];

  // ---------------- state ---------------------------------------------------
  float x0 = xz0[(size_t)b * NXZ + 0];
  float x1 = xz0[(size_t)b * NXZ + 1];
  float x2;                               // = Cc0 = eC(z0), set below
  float z[15];
#pragma unroll
  for (int j = 0; j < 15; ++j) z[j] = xz0[(size_t)b * NXZ + 3 + j];

  const float* up = u + (size_t)b * T_STEPS;
  float* o1 = out + (size_t)b * T_STEPS * 4;                                   // yseq_full
  float* o2 = out + (size_t)B_ELEMS * T_STEPS * 4 + (size_t)b * T_STEPS * 2;   // yseq
  float* o3 = out + (size_t)B_ELEMS * T_STEPS * 6 + (size_t)b * T_STEPS * 3;   // xseq

  // ---------------- eC(z) for the wave's 32 elements via 4 WMMAs -----------
  auto evalCc = [&]() -> float {
    // pack this lane's z to f16 pairs: dword i = halves (2i, 2i+1) <-> K=2i,2i+1
    unsigned int zp[8];
#pragma unroll
    for (int i = 0; i < 8; ++i)
      zp[i] = pk2h(z[2 * i], (2 * i + 1 < 15) ? z[2 * i + 1] : 0.f);
    // B tile0: lanes 0-15 own elements 0-15 (K=0..15); lanes>=16 carry K=16..31 = pad
    // B tile1: lanes 0-15 need element (lane+16)'s z -> one packed shfl_xor(16) per dword
    v8u b0u = {}, b1u = {};
#pragma unroll
    for (int i = 0; i < 8; ++i) {
      unsigned int p = zp[i];
      unsigned int q = (unsigned int)__shfl_xor((int)p, 16);
      b0u[i] = lolane ? p : 0u;
      b1u[i] = lolane ? q : 0u;
    }
    v16h B0 = __builtin_bit_cast(v16h, b0u);
    v16h B1 = __builtin_bit_cast(v16h, b1u);
    v8f cz = {};
    v8f c00 = __builtin_amdgcn_wmma_f32_16x16x32_f16(false, A0, false, B0, (short)0, cz, false, false);
    v8f c01 = __builtin_amdgcn_wmma_f32_16x16x32_f16(false, A1, false, B0, (short)0, cz, false, false);
    v8f c10 = __builtin_amdgcn_wmma_f32_16x16x32_f16(false, A0, false, B1, (short)0, cz, false, false);
    v8f c11 = __builtin_amdgcn_wmma_f32_16x16x32_f16(false, A1, false, B1, (short)0, cz, false, false);
    // layer 2: tanh + dot over hidden; 4 independent 8-deep chains
    float s0a = 0.f, s0b = 0.f, s1a = 0.f, s1b = 0.f;
#pragma unroll
    for (int r = 0; r < 8; ++r) {
      s0a = fmaf(fast_tanh(c00[r] + eb1A[r]), ew2A[r], s0a);
      s0b = fmaf(fast_tanh(c01[r] + eb1B[r]), ew2B[r], s0b);
      s1a = fmaf(fast_tanh(c10[r] + eb1A[r]), ew2A[r], s1a);
      s1b = fmaf(fast_tanh(c11[r] + eb1B[r]), ew2B[r], s1b);
    }
    float s0 = s0a + s0b;   // element tile 0, this lane's partial
    float s1 = s1a + s1b;   // element tile 1, this lane's partial
    // single exchange: peer needs s0 from upper half / s1 from lower half
    float send = lolane ? s1 : s0;
    float recv = __shfl_xor(send, 16);
    return (lolane ? s0 : s1) + recv + eb2;
  };

  // ---------------- fxu: RK4 stage derivative ------------------------------
  auto fxu = [&](float a, float bb, float c, float uf,
                 float& d0, float& d1, float& d2) {
    // split each dot into even/odd accumulators: chain depth 8 instead of 16
    float a1e = 0.f, a1o = 0.f, a2e = 0.f, a2o = 0.f;
#pragma unroll
    for (int h = 0; h < 8; ++h) {
      const int he = 2 * h, ho = 2 * h + 1;
      a1e = fmaf(fast_tanh(fmaf(a,  r1w1[he],  r1b1v[he])), r1w2[he], a1e);
      a1o = fmaf(fast_tanh(fmaf(a,  r1w1[ho],  r1b1v[ho])), r1w2[ho], a1o);
      a2e = fmaf(fast_tanh(fmaf(bb, r2w1a[he], fmaf(c, r2w1b[he], r2b1v[he]))), r2w2[he], a2e);
      a2o = fmaf(fast_tanh(fmaf(bb, r2w1a[ho], fmaf(c, r2w1b[ho], r2b1v[ho]))), r2w2[ho], a2o);
    }
    float rr1 = (r1b2v + a1e + a1o) * 0.3f;   // * CASTD
    float rr2 = (r2b2v + a2e + a2o) * 0.2f;   // * CBSTD
    float Ca  = fmaf(a,  0.3f, 0.5f);
    float Cb  = fmaf(bb, 0.2f, 0.4f);
    float Cc  = fmaf(c,  0.2f, 0.4f);
    float Caf = fmaf(uf, 0.5f, 1.0f);
    float dCa = 0.1f * (Caf - Ca) - rr1;          // F/V = 0.1
    float dCb = -0.1f * Cb + rr1 - 3.0f * rr2;
    float dCc = -0.1f * Cc + rr2;
    d0 = dCa * (1.0f / 0.3f);
    d1 = dCb * (1.0f / 0.2f);
    d2 = dCc * (1.0f / 0.2f);
  };

  // peeled: x2(0) = Cc0 = eC(z0)  (no per-iteration t==0 check)
  x2 = evalCc();

  // ---------------- sequential scan ----------------------------------------
  for (int t = 0; t < T_STEPS; ++t) {
    float CcNN = evalCc();               // off the recurrence critical path

    // ---- emit y_t = (x0, x1, x2, CcNN) (state BEFORE update) ----
    *reinterpret_cast<float4*>(o1 + 4 * (size_t)t) = make_float4(x0, x1, x2, CcNN);
    *reinterpret_cast<float2*>(o2 + 2 * (size_t)t) = make_float2(x0, x1);
    o3[3 * (size_t)t + 0] = x0;
    o3[3 * (size_t)t + 1] = x1;
    o3[3 * (size_t)t + 2] = x2;

    // ---- input + prefetch next cacheline of this lane's u stream ----
    if (((t & 31) == 0) && (t + 64 < T_STEPS))
      __builtin_prefetch(up + t + 64, 0, 0);
    float uu = up[t];

    // ---- RK4 (DELTA = 1) ----
    float k10, k11, k12, k20, k21, k22, k30, k31, k32, k40, k41, k42;
    fxu(x0, x1, x2, uu, k10, k11, k12);
    fxu(x0 + 0.5f * k10, x1 + 0.5f * k11, x2 + 0.5f * k12, uu, k20, k21, k22);
    fxu(x0 + 0.5f * k20, x1 + 0.5f * k21, x2 + 0.5f * k22, uu, k30, k31, k32);
    fxu(x0 + k30, x1 + k31, x2 + k32, uu, k40, k41, k42);
    float nx0 = x0 + (1.0f / 6.0f) * ((k10 + k40) + 2.f * (k20 + k30));
    float nx1 = x1 + (1.0f / 6.0f) * ((k11 + k41) + 2.f * (k21 + k31));
    float nx2 = x2 + (1.0f / 6.0f) * ((k12 + k42) + 2.f * (k22 + k32));

    // ---- shift register (uses OLD x0,x1) ----
    float zn[15] = { z[2], z[3], z[4], z[5], z[6], z[7], z[8], z[9],
                     x0, x1,
                     z[11], z[12], z[13], z[14],
                     uu };
#pragma unroll
    for (int j = 0; j < 15; ++j) z[j] = zn[j];

    x0 = nx0; x1 = nx1; x2 = nx2;
  }
}

extern "C" void kernel_launch(void* const* d_in, const int* in_sizes, int n_in,
                              void* d_out, int out_size, void* d_ws, size_t ws_size,
                              hipStream_t stream) {
  (void)in_sizes; (void)n_in; (void)out_size; (void)d_ws; (void)ws_size;
  const float* u     = (const float*)d_in[0];
  const float* xz0   = (const float*)d_in[1];
  const float* r1_W1 = (const float*)d_in[2];
  const float* r1_b1 = (const float*)d_in[3];
  const float* r1_W2 = (const float*)d_in[4];
  const float* r1_b2 = (const float*)d_in[5];
  const float* r2_W1 = (const float*)d_in[6];
  const float* r2_b1 = (const float*)d_in[7];
  const float* r2_W2 = (const float*)d_in[8];
  const float* r2_b2 = (const float*)d_in[9];
  const float* eC_W1 = (const float*)d_in[10];
  const float* eC_b1 = (const float*)d_in[11];
  const float* eC_W2 = (const float*)d_in[12];
  const float* eC_b2 = (const float*)d_in[13];
  float* out = (float*)d_out;

  dim3 grid(B_ELEMS / 32), block(32);   // one wave32 per block, lane = batch element
  reac_scan_kernel<<<grid, block, 0, stream>>>(u, xz0,
                                               r1_W1, r1_b1, r1_W2, r1_b2,
                                               r2_W1, r2_b1, r2_W2, r2_b2,
                                               eC_W1, eC_b1, eC_W2, eC_b2,
                                               out);
}
